// SimpleGVPLayer_16887811407937
// MI455X (gfx1250) — matrix operand order repaired and checked by reading, not compiled, not run on hardware
//
#include <hip/hip_runtime.h>
#include <math.h>

typedef __attribute__((ext_vector_type(16))) __bf16 v16bf;
typedef __attribute__((ext_vector_type(8)))  __bf16 v8bf;
typedef __attribute__((ext_vector_type(4)))  __bf16 v4bf;
typedef __attribute__((ext_vector_type(8)))  float  v8f;

#define NNODE 20000
#define NEDGE 640000
#define SD 128
#define VD 16
#define ED 32

#define MSG_K   322
#define MSG_KP  352   // padded to multiple of 32
#define UPD_K   288   // exactly 9*32
#define MSG_VK  33
#define MSG_VKP 64
#define UPD_VK  32

// ---- workspace layout (bytes, all 256-aligned) ----
#define OFF_MSGW1 0
#define OFF_MSGW2 (OFF_MSGW1 + MSG_KP*128*2)          //  90112
#define OFF_MSGWG (OFF_MSGW2 + 128*128*2)             // 122880
#define OFF_MSGWV (OFF_MSGWG + 128*16*2)              // 126976
#define OFF_UPDW1 (OFF_MSGWV + MSG_VKP*16*2)          // 129024
#define OFF_UPDW2 (OFF_UPDW1 + UPD_K*128*2)           // 202752
#define OFF_UPDWG (OFF_UPDW2 + 128*128*2)             // 235520
#define OFF_UPDWV (OFF_UPDWG + 128*16*2)              // 239616
#define OFF_AGGS  (OFF_UPDWV + UPD_VK*16*2)           // 240640
#define OFF_AGGV  (OFF_AGGS + (size_t)NNODE*128*4)

#define WMMA_BF16(A,B,C) \
  __builtin_amdgcn_wmma_f32_16x16x32_bf16(false,(A),false,(B),(short)0,(C),false,false)

// A-fragment from LDS: 16x32 bf16 tile, row M = lane&15, split-K per ISA layout.
__device__ __forceinline__ v16bf load_a_lds(const __bf16* base, int rowStride,
                                            int kt, int lane) {
  const int m  = lane & 15;
  const int hi = (lane >> 4) & 1;
  const __bf16* p = base + m * rowStride + kt * 32 + hi * 8;
  union { v16bf v; v8bf h[2]; } u;
  u.h[0] = *(const v8bf*)(p);        // K = base .. base+7
  u.h[1] = *(const v8bf*)(p + 16);   // K = base+16 .. base+23
  return u.v;
}

// B-fragment from global: 32x16 bf16 tile, col N = lane&15, split-K per ISA layout.
__device__ __forceinline__ v16bf load_b_glb(const __bf16* W, int ncols,
                                            int kt, int n0, int lane) {
  const int n  = lane & 15;
  const int hi = (lane >> 4) & 1;
  const __bf16* p = W + (size_t)(kt * 32 + hi * 8) * ncols + n0 + n;
  v16bf b;
#pragma unroll
  for (int s = 0; s < 8; ++s) {
    b[s]     = p[(size_t)s * ncols];
    b[8 + s] = p[(size_t)(16 + s) * ncols];
  }
  return b;
}

__device__ __forceinline__ void st4bf(__bf16* p, float4 f) {
  v4bf t;
  t[0] = (__bf16)f.x; t[1] = (__bf16)f.y; t[2] = (__bf16)f.z; t[3] = (__bf16)f.w;
  *(v4bf*)p = t;
}

__device__ __forceinline__ float sigmoidf_(float x) {
  return 1.0f / (1.0f + __expf(-x));
}

// ---------------------------------------------------------------------------
__global__ void zero_kernel(float* p, size_t n) {
  size_t i = (size_t)blockIdx.x * blockDim.x + threadIdx.x;
  size_t stride = (size_t)gridDim.x * blockDim.x;
  for (; i < n; i += stride) p[i] = 0.0f;
}

__global__ void cvt_pad_kernel(const float* __restrict__ src, __bf16* __restrict__ dst,
                               int srcRows, int dstRows, int cols) {
  int i = blockIdx.x * blockDim.x + threadIdx.x;
  int total = dstRows * cols;
  if (i < total) {
    int r = i / cols;
    dst[i] = (r < srcRows) ? (__bf16)src[i] : (__bf16)0.0f;
  }
}

// ---------------------------------------------------------------------------
// Per-edge message GVP + scatter-add.
// 1 wave per block; one 32-edge tile per wave (M = 2x16 so each B-fragment
// feeds two WMMAs).
__global__ __launch_bounds__(32) void edge_msg_kernel(
    const float* __restrict__ s, const float* __restrict__ v,
    const int* __restrict__ eidx, const float* __restrict__ es,
    const float* __restrict__ ev,
    const __bf16* __restrict__ W1, const float* __restrict__ b1,
    const __bf16* __restrict__ W2, const float* __restrict__ b2,
    const __bf16* __restrict__ Wv, const __bf16* __restrict__ Wg,
    const float* __restrict__ bg,
    float* __restrict__ aggs, float* __restrict__ aggv)
{
  __shared__ __bf16 smem[32 * MSG_KP + 3 * 32 * MSG_VKP];  // 34816 bytes
  const int lane = threadIdx.x & 31;
  const int e0   = blockIdx.x * 32;

  __bf16* X   = smem;                 // [32][352]
  __bf16* VIN = smem + 32 * MSG_KP;   // [3][32][64]

  // ---------------- gather: each lane owns one edge ----------------
  {
    const int e  = lane;
    const int eg = e0 + e;
    const int src = eidx[eg];
    const int dst = eidx[NEDGE + eg];
    const float4* sr = (const float4*)(s + (size_t)src * SD);
    const float4* sd = (const float4*)(s + (size_t)dst * SD);
    __bf16* Xe = X + e * MSG_KP;
#pragma unroll 8
    for (int i = 0; i < 32; ++i) {
      st4bf(Xe + i * 4, sr[i]);
      st4bf(Xe + 128 + i * 4, sd[i]);
    }
    const float4* ep = (const float4*)(es + (size_t)eg * ED);
#pragma unroll
    for (int i = 0; i < 8; ++i) st4bf(Xe + 256 + i * 4, ep[i]);
#pragma unroll
    for (int w = 0; w < 16; ++w) {
      const float* vp = v + (size_t)src * (VD * 3) + w * 3;
      float x = vp[0], y = vp[1], z = vp[2];
      Xe[289 + w] = (__bf16)sqrtf(fmaxf(x * x + y * y + z * z, 1e-8f));
      VIN[0 * 32 * MSG_VKP + e * MSG_VKP + w] = (__bf16)x;
      VIN[1 * 32 * MSG_VKP + e * MSG_VKP + w] = (__bf16)y;
      VIN[2 * 32 * MSG_VKP + e * MSG_VKP + w] = (__bf16)z;
      const float* vq = v + (size_t)dst * (VD * 3) + w * 3;
      float x2 = vq[0], y2 = vq[1], z2 = vq[2];
      Xe[305 + w] = (__bf16)sqrtf(fmaxf(x2 * x2 + y2 * y2 + z2 * z2, 1e-8f));
      VIN[0 * 32 * MSG_VKP + e * MSG_VKP + 16 + w] = (__bf16)x2;
      VIN[1 * 32 * MSG_VKP + e * MSG_VKP + 16 + w] = (__bf16)y2;
      VIN[2 * 32 * MSG_VKP + e * MSG_VKP + 16 + w] = (__bf16)z2;
    }
    {
      const float* evp = ev + (size_t)eg * 3;
      float x = evp[0], y = evp[1], z = evp[2];
      float nrm = sqrtf(fmaxf(x * x + y * y + z * z, 1e-8f));
      Xe[288] = (__bf16)nrm;   // edge_len
      Xe[321] = (__bf16)nrm;   // vn of edge_v
      VIN[0 * 32 * MSG_VKP + e * MSG_VKP + 32] = (__bf16)x;
      VIN[1 * 32 * MSG_VKP + e * MSG_VKP + 32] = (__bf16)y;
      VIN[2 * 32 * MSG_VKP + e * MSG_VKP + 32] = (__bf16)z;
    }
    for (int i = 322; i < MSG_KP; ++i) Xe[i] = (__bf16)0.0f;
#pragma unroll
    for (int c = 0; c < 3; ++c)
      for (int i = 33; i < MSG_VKP; ++i)
        VIN[c * 32 * MSG_VKP + e * MSG_VKP + i] = (__bf16)0.0f;
  }
  __syncthreads();   // single-wave WG: free; keeps LDS ordering explicit

  const int col = lane & 15;
  const int hi  = lane >> 4;

  // ---------------- h = X @ W1 + b1 ----------------
  v8f acc[2][8];
#pragma unroll
  for (int nt = 0; nt < 8; ++nt) {
    float bv = b1[nt * 16 + col];
    v8f a;
#pragma unroll
    for (int r = 0; r < 8; ++r) a[r] = bv;
    acc[0][nt] = a; acc[1][nt] = a;
  }
  for (int kt = 0; kt < 11; ++kt) {
    v16bf a0 = load_a_lds(X, MSG_KP, kt, lane);
    v16bf a1 = load_a_lds(X + 16 * MSG_KP, MSG_KP, kt, lane);
#pragma unroll
    for (int nt = 0; nt < 8; ++nt) {
      v16bf bf = load_b_glb(W1, 128, kt, nt * 16, lane);
      acc[0][nt] = WMMA_BF16(a0, bf, acc[0][nt]);
      acc[1][nt] = WMMA_BF16(a1, bf, acc[1][nt]);
    }
  }

  // ---------------- SiLU -> X2 (reuse X, [32][128]) ----------------
  __bf16* X2 = X;
#pragma unroll
  for (int mt = 0; mt < 2; ++mt)
#pragma unroll
    for (int nt = 0; nt < 8; ++nt)
#pragma unroll
      for (int r = 0; r < 8; ++r) {
        float h = acc[mt][nt][r];
        X2[(mt * 16 + r + 8 * hi) * 128 + nt * 16 + col] = (__bf16)(h * sigmoidf_(h));
      }

  // ---------------- s_out = silu(h) @ W2 + b2 ----------------
  v8f acc2[2][8];
#pragma unroll
  for (int nt = 0; nt < 8; ++nt) {
    float bv = b2[nt * 16 + col];
    v8f a;
#pragma unroll
    for (int r = 0; r < 8; ++r) a[r] = bv;
    acc2[0][nt] = a; acc2[1][nt] = a;
  }
  for (int kt = 0; kt < 4; ++kt) {
    v16bf a0 = load_a_lds(X2, 128, kt, lane);
    v16bf a1 = load_a_lds(X2 + 16 * 128, 128, kt, lane);
#pragma unroll
    for (int nt = 0; nt < 8; ++nt) {
      v16bf bf = load_b_glb(W2, 128, kt, nt * 16, lane);
      acc2[0][nt] = WMMA_BF16(a0, bf, acc2[0][nt]);
      acc2[1][nt] = WMMA_BF16(a1, bf, acc2[1][nt]);
    }
  }

  // stage s_out (bf16) for gate matmul (reuse X region; W2 A-loads done)
  __bf16* X3 = X;
#pragma unroll
  for (int mt = 0; mt < 2; ++mt)
#pragma unroll
    for (int nt = 0; nt < 8; ++nt)
#pragma unroll
      for (int r = 0; r < 8; ++r)
        X3[(mt * 16 + r + 8 * hi) * 128 + nt * 16 + col] = (__bf16)acc2[mt][nt][r];

  // dst node ids for scatter
  int dstm[2][8];
#pragma unroll
  for (int mt = 0; mt < 2; ++mt)
#pragma unroll
    for (int r = 0; r < 8; ++r)
      dstm[mt][r] = eidx[NEDGE + e0 + mt * 16 + r + 8 * hi];

  // scatter m_s
#pragma unroll
  for (int mt = 0; mt < 2; ++mt)
#pragma unroll
    for (int nt = 0; nt < 8; ++nt)
#pragma unroll
      for (int r = 0; r < 8; ++r)
        atomicAdd(aggs + (size_t)dstm[mt][r] * 128 + nt * 16 + col, acc2[mt][nt][r]);

  // ---------------- gate = sigmoid(s_out @ Wg + bg) ----------------
  v8f gacc[2];
  {
    float bv = bg[col];
#pragma unroll
    for (int r = 0; r < 8; ++r) { gacc[0][r] = bv; gacc[1][r] = bv; }
  }
  for (int kt = 0; kt < 4; ++kt) {
    v16bf a0 = load_a_lds(X3, 128, kt, lane);
    v16bf a1 = load_a_lds(X3 + 16 * 128, 128, kt, lane);
    v16bf bf = load_b_glb(Wg, 16, kt, 0, lane);
    gacc[0] = WMMA_BF16(a0, bf, gacc[0]);
    gacc[1] = WMMA_BF16(a1, bf, gacc[1]);
  }
  float gate[2][8];
#pragma unroll
  for (int mt = 0; mt < 2; ++mt)
#pragma unroll
    for (int r = 0; r < 8; ++r) gate[mt][r] = sigmoidf_(gacc[mt][r]);

  // ---------------- v_proj = VIN @ Wv ; m_v = v_proj * gate ----------------
#pragma unroll
  for (int c = 0; c < 3; ++c) {
    v8f vac[2];
#pragma unroll
    for (int r = 0; r < 8; ++r) { vac[0][r] = 0.0f; vac[1][r] = 0.0f; }
    const __bf16* vinc = VIN + c * 32 * MSG_VKP;
#pragma unroll
    for (int kt = 0; kt < 2; ++kt) {
      v16bf a0 = load_a_lds(vinc, MSG_VKP, kt, lane);
      v16bf a1 = load_a_lds(vinc + 16 * MSG_VKP, MSG_VKP, kt, lane);
      v16bf bf = load_b_glb(Wv, 16, kt, 0, lane);
      vac[0] = WMMA_BF16(a0, bf, vac[0]);
      vac[1] = WMMA_BF16(a1, bf, vac[1]);
    }
#pragma unroll
    for (int mt = 0; mt < 2; ++mt)
#pragma unroll
      for (int r = 0; r < 8; ++r)
        atomicAdd(aggv + (size_t)dstm[mt][r] * 48 + col * 3 + c,
                  vac[mt][r] * gate[mt][r]);
  }
}

// ---------------------------------------------------------------------------
// Node update GVP + residual + LayerNorm.  2 waves/block, 16 nodes per wave.
__global__ __launch_bounds__(64) void node_upd_kernel(
    const float* __restrict__ s, const float* __restrict__ v,
    const float* __restrict__ aggs, const float* __restrict__ aggv,
    const __bf16* __restrict__ W1, const float* __restrict__ b1,
    const __bf16* __restrict__ W2, const float* __restrict__ b2,
    const __bf16* __restrict__ Wv, const __bf16* __restrict__ Wg,
    const float* __restrict__ bg,
    const float* __restrict__ ln_g, const float* __restrict__ ln_b,
    float* __restrict__ out_s, float* __restrict__ out_v)
{
  __shared__ __bf16 nsmem[2 * (16 * UPD_K + 3 * 16 * UPD_VK)];
  __shared__ float  snmem[2 * 16 * 128];
  const int lane = threadIdx.x & 31;
  const int wave = threadIdx.x >> 5;
  const int tile = blockIdx.x * 2 + wave;
  const int n0   = tile * 16;

  __bf16* X   = nsmem + wave * (16 * UPD_K + 3 * 16 * UPD_VK); // [16][288]
  __bf16* VIN = X + 16 * UPD_K;                                 // [3][16][32]
  float*  SN  = snmem + wave * 16 * 128;                        // [16][128]

  // ---------------- gather ----------------
  {
    const int e = lane >> 1, half = lane & 1;
    const int node = n0 + e;
    const float4* sp = (const float4*)(s + (size_t)node * SD);
    const float4* ap = (const float4*)(aggs + (size_t)node * SD);
    __bf16* Xe = X + e * UPD_K;
#pragma unroll 4
    for (int i = 0; i < 16; ++i) {
      st4bf(Xe + half * 64 + i * 4, sp[half * 16 + i]);
      st4bf(Xe + 128 + half * 64 + i * 4, ap[half * 16 + i]);
    }
#pragma unroll
    for (int w = 0; w < 8; ++w) {
      int wi = half * 8 + w;
      const float* vp = v + (size_t)node * (VD * 3) + wi * 3;
      float x = vp[0], y = vp[1], z = vp[2];
      Xe[256 + wi] = (__bf16)sqrtf(fmaxf(x * x + y * y + z * z, 1e-8f));
      VIN[0 * 16 * UPD_VK + e * UPD_VK + wi] = (__bf16)x;
      VIN[1 * 16 * UPD_VK + e * UPD_VK + wi] = (__bf16)y;
      VIN[2 * 16 * UPD_VK + e * UPD_VK + wi] = (__bf16)z;
      const float* aq = aggv + (size_t)node * (VD * 3) + wi * 3;
      float x2 = aq[0], y2 = aq[1], z2 = aq[2];
      Xe[272 + wi] = (__bf16)sqrtf(fmaxf(x2 * x2 + y2 * y2 + z2 * z2, 1e-8f));
      VIN[0 * 16 * UPD_VK + e * UPD_VK + 16 + wi] = (__bf16)x2;
      VIN[1 * 16 * UPD_VK + e * UPD_VK + 16 + wi] = (__bf16)y2;
      VIN[2 * 16 * UPD_VK + e * UPD_VK + 16 + wi] = (__bf16)z2;
    }
  }
  __syncthreads();

  const int col = lane & 15;
  const int hi  = lane >> 4;

  // ---------------- h = X @ W1 + b1 ----------------
  v8f acc[8];
#pragma unroll
  for (int nt = 0; nt < 8; ++nt) {
    float bv = b1[nt * 16 + col];
    v8f a;
#pragma unroll
    for (int r = 0; r < 8; ++r) a[r] = bv;
    acc[nt] = a;
  }
  for (int kt = 0; kt < 9; ++kt) {
    v16bf af = load_a_lds(X, UPD_K, kt, lane);
#pragma unroll
    for (int nt = 0; nt < 8; ++nt) {
      v16bf bf = load_b_glb(W1, 128, kt, nt * 16, lane);
      acc[nt] = WMMA_BF16(af, bf, acc[nt]);
    }
  }

  __bf16* X2 = X;
#pragma unroll
  for (int nt = 0; nt < 8; ++nt)
#pragma unroll
    for (int r = 0; r < 8; ++r) {
      float h = acc[nt][r];
      X2[(r + 8 * hi) * 128 + nt * 16 + col] = (__bf16)(h * sigmoidf_(h));
    }

  // ---------------- ds = silu(h) @ W2 + b2 ----------------
  v8f acc2[8];
#pragma unroll
  for (int nt = 0; nt < 8; ++nt) {
    float bv = b2[nt * 16 + col];
    v8f a;
#pragma unroll
    for (int r = 0; r < 8; ++r) a[r] = bv;
    acc2[nt] = a;
  }
  for (int kt = 0; kt < 4; ++kt) {
    v16bf af = load_a_lds(X2, 128, kt, lane);
#pragma unroll
    for (int nt = 0; nt < 8; ++nt) {
      v16bf bf = load_b_glb(W2, 128, kt, nt * 16, lane);
      acc2[nt] = WMMA_BF16(af, bf, acc2[nt]);
    }
  }

  // stage ds (bf16) for gate matmul; stage s + ds (fp32) for layernorm
  __bf16* X3 = X;
#pragma unroll
  for (int nt = 0; nt < 8; ++nt)
#pragma unroll
    for (int r = 0; r < 8; ++r) {
      int m = r + 8 * hi, c = nt * 16 + col;
      float ds = acc2[nt][r];
      X3[m * 128 + c] = (__bf16)ds;
      SN[m * 128 + c] = s[(size_t)(n0 + m) * 128 + c] + ds;
    }

  // ---------------- gate ----------------
  v8f gacc;
  {
    float bv = bg[col];
#pragma unroll
    for (int r = 0; r < 8; ++r) gacc[r] = bv;
  }
  for (int kt = 0; kt < 4; ++kt) {
    v16bf af = load_a_lds(X3, 128, kt, lane);
    v16bf bf = load_b_glb(Wg, 16, kt, 0, lane);
    gacc = WMMA_BF16(af, bf, gacc);
  }
  float gate[8];
#pragma unroll
  for (int r = 0; r < 8; ++r) gate[r] = sigmoidf_(gacc[r]);

  // ---------------- v_out = v + (VIN @ Wv) * gate ----------------
#pragma unroll
  for (int c = 0; c < 3; ++c) {
    v8f vac;
#pragma unroll
    for (int r = 0; r < 8; ++r) vac[r] = 0.0f;
    v16bf af = load_a_lds(VIN + c * 16 * UPD_VK, UPD_VK, 0, lane);
    v16bf bf = load_b_glb(Wv, 16, 0, 0, lane);
    vac = WMMA_BF16(af, bf, vac);
#pragma unroll
    for (int r = 0; r < 8; ++r) {
      int node = n0 + r + 8 * hi;
      size_t o = (size_t)node * 48 + col * 3 + c;
      out_v[o] = v[o] + vac[r] * gate[r];
    }
  }

  __syncthreads();

  // ---------------- LayerNorm(s + ds) ----------------
  if (lane < 16) {
    int row = lane, node = n0 + row;
    const float* sr = SN + row * 128;
    float mu = 0.0f;
    for (int i = 0; i < 128; ++i) mu += sr[i];
    mu *= (1.0f / 128.0f);
    float var = 0.0f;
    for (int i = 0; i < 128; ++i) { float d = sr[i] - mu; var += d * d; }
    var *= (1.0f / 128.0f);
    float rstd = rsqrtf(var + 1e-5f);
    float* op = out_s + (size_t)node * 128;
    for (int i = 0; i < 128; ++i)
      op[i] = (sr[i] - mu) * rstd * ln_g[i] + ln_b[i];
  }
}

// ---------------------------------------------------------------------------
extern "C" void kernel_launch(void* const* d_in, const int* in_sizes, int n_in,
                              void* d_out, int out_size, void* d_ws, size_t ws_size,
                              hipStream_t stream) {
  (void)in_sizes; (void)n_in; (void)out_size; (void)ws_size;
  const float* s       = (const float*)d_in[0];
  const float* v       = (const float*)d_in[1];
  const int*   eidx    = (const int*)d_in[2];
  const float* es      = (const float*)d_in[3];
  const float* ev      = (const float*)d_in[4];
  const float* msg_W1  = (const float*)d_in[5];
  const float* msg_b1  = (const float*)d_in[6];
  const float* msg_W2  = (const float*)d_in[7];
  const float* msg_b2  = (const float*)d_in[8];
  const float* msg_Wv  = (const float*)d_in[9];
  const float* msg_Wg  = (const float*)d_in[10];
  const float* msg_bg  = (const float*)d_in[11];
  const float* upd_W1  = (const float*)d_in[12];
  const float* upd_b1  = (const float*)d_in[13];
  const float* upd_W2  = (const float*)d_in[14];
  const float* upd_b2  = (const float*)d_in[15];
  const float* upd_Wv  = (const float*)d_in[16];
  const float* upd_Wg  = (const float*)d_in[17];
  const float* upd_bg  = (const float*)d_in[18];
  const float* ln_g    = (const float*)d_in[19];
  const float* ln_b    = (const float*)d_in[20];

  char* ws = (char*)d_ws;
  __bf16* mW1 = (__bf16*)(ws + OFF_MSGW1);
  __bf16* mW2 = (__bf16*)(ws + OFF_MSGW2);
  __bf16* mWg = (__bf16*)(ws + OFF_MSGWG);
  __bf16* mWv = (__bf16*)(ws + OFF_MSGWV);
  __bf16* uW1 = (__bf16*)(ws + OFF_UPDW1);
  __bf16* uW2 = (__bf16*)(ws + OFF_UPDW2);
  __bf16* uWg = (__bf16*)(ws + OFF_UPDWG);
  __bf16* uWv = (__bf16*)(ws + OFF_UPDWV);
  float*  aggs = (float*)(ws + OFF_AGGS);
  float*  aggv = (float*)(ws + OFF_AGGV);

  float* out_s = (float*)d_out;
  float* out_v = out_s + (size_t)NNODE * 128;

  // zero aggregation buffers (aggs & aggv are contiguous)
  zero_kernel<<<2048, 256, 0, stream>>>(aggs, (size_t)NNODE * (128 + 48));

  // weights -> bf16 (with zero row padding)
  auto cvt = [&](const float* src, __bf16* dst, int sr, int dr, int c) {
    int total = dr * c;
    cvt_pad_kernel<<<(total + 255) / 256, 256, 0, stream>>>(src, dst, sr, dr, c);
  };
  cvt(msg_W1, mW1, MSG_K, MSG_KP, 128);
  cvt(msg_W2, mW2, 128, 128, 128);
  cvt(msg_Wv, mWv, MSG_VK, MSG_VKP, 16);
  cvt(msg_Wg, mWg, 128, 128, 16);
  cvt(upd_W1, uW1, UPD_K, UPD_K, 128);
  cvt(upd_W2, uW2, 128, 128, 128);
  cvt(upd_Wv, uWv, UPD_VK, UPD_VK, 16);
  cvt(upd_Wg, uWg, 128, 128, 16);

  // per-edge messages + scatter (20000 tiles of 32 edges, 1 wave/block)
  edge_msg_kernel<<<NEDGE / 32, 32, 0, stream>>>(
      s, v, eidx, es, ev, mW1, msg_b1, mW2, msg_b2, mWv, mWg, msg_bg, aggs, aggv);

  // node update + layernorm (1250 tiles, 2 waves/block)
  node_upd_kernel<<<NNODE / 32, 64, 0, stream>>>(
      s, v, aggs, aggv, uW1, upd_b1, uW2, upd_b2, uWv, uWg, upd_bg,
      ln_g, ln_b, out_s, out_v);
}
